// HierarchicalDecoder_67963562492642
// MI455X (gfx1250) — matrix operand order repaired and checked by reading, not compiled, not run on hardware
//
#include <hip/hip_runtime.h>

// ---------------------------------------------------------------------------
// Types
// ---------------------------------------------------------------------------
typedef __attribute__((ext_vector_type(16))) __bf16 v16bf;
typedef __attribute__((ext_vector_type(8)))  float  v8f;
typedef __attribute__((ext_vector_type(4)))  unsigned int v4u;
typedef __attribute__((ext_vector_type(4)))  int v4i;

#ifndef __has_builtin
#define __has_builtin(x) 0
#endif

#if __has_builtin(__builtin_amdgcn_global_load_async_to_lds_b128)
#define HAVE_ASYNC_LDS 1
#else
#define HAVE_ASYNC_LDS 0
#endif

#if __has_builtin(__builtin_amdgcn_s_wait_asynccnt)
#define HAVE_WAIT_ASYNC 1
#else
#define HAVE_WAIT_ASYNC 0
#endif

// ---------------------------------------------------------------------------
// Tiling constants
// ---------------------------------------------------------------------------
#define MT 128              // rows (patients) per block
#define NT 64               // cols (classes) per block
#define KT 32               // K slice per stage (bf16 WMMA K)
#define NKSTEP 16           // 512 / 32
#define A_ROW_STRIDE 80     // 64B of K data + 16B pad (bank spread)
#define A_PLANE (128 * A_ROW_STRIDE)       // 10240 B per hi/lo plane of A
#define B_PLANE (64 * A_ROW_STRIDE)        // 5120 B per hi/lo plane of B
#define B_BASE  (2 * A_PLANE)              // 20480
#define BUF_BYTES (B_BASE + 2 * B_PLANE)   // 30720 per stage buffer
#define CHUNKS_PER_THREAD 6                // 1536 x 16B chunks / 256 threads

template <int N>
__device__ __forceinline__ void wait_asynccnt_le() {
#if HAVE_ASYNC_LDS
#if HAVE_WAIT_ASYNC
  __builtin_amdgcn_s_wait_asynccnt(N);
#else
  asm volatile("s_wait_asynccnt %0" ::"i"(N) : "memory");
#endif
#endif
}

union Frag {
  v4u   q[2];   // two 16B halves, VGPR order
  v16bf v;      // WMMA operand
};

// ---------------------------------------------------------------------------
// fp32 -> bf16 hi/lo split of the patient embedding (row-major [4096][512])
// ---------------------------------------------------------------------------
__global__ void hd_split_x(const float* __restrict__ X,
                           __bf16* __restrict__ Xh, __bf16* __restrict__ Xl,
                           int n) {
  int i = blockIdx.x * blockDim.x + threadIdx.x;
  if (i < n) {
    float v  = X[i];
    __bf16 h = (__bf16)v;
    Xh[i] = h;
    Xl[i] = (__bf16)(v - (float)h);
  }
}

// ---------------------------------------------------------------------------
// W [512][N] (N contiguous) -> K-major padded Wt_hi/lo [Npad][512], zero fill
// ---------------------------------------------------------------------------
__global__ void hd_split_w(const float* __restrict__ W,
                           __bf16* __restrict__ Wh, __bf16* __restrict__ Wl,
                           int N) {
  __shared__ float t[32][33];
  const int k0 = blockIdx.y * 32;
  const int n0 = blockIdx.x * 32;
  // load: x over n (coalesced global read), rows over k
  for (int i = threadIdx.y; i < 32; i += 8) {
    int k = k0 + i, n = n0 + threadIdx.x;
    t[i][threadIdx.x] = (n < N) ? W[(size_t)k * N + n] : 0.0f;
  }
  __syncthreads();
  // store: x over k (coalesced K-major write), rows over n
  for (int i = threadIdx.y; i < 32; i += 8) {
    int n = n0 + i, k = k0 + threadIdx.x;
    float v  = t[threadIdx.x][i];
    __bf16 h = (__bf16)v;
    Wh[(size_t)n * 512 + k] = h;
    Wl[(size_t)n * 512 + k] = (__bf16)(v - (float)h);
  }
}

// ---------------------------------------------------------------------------
// Fused level GEMM: out[b, j] = sigmoid(X@W + bias)[b, j] * prev[b, par[j]]
// 128x64 tile / block, 8 waves, each wave a 32x32 sub-tile (2x2 WMMA tiles),
// split-bf16 (hi*hi + hi*lo + lo*hi) with f32 accumulation.
// ---------------------------------------------------------------------------
template <bool HASP>
__global__ __launch_bounds__(256) void hd_gemm(
    const __bf16* __restrict__ Xh, const __bf16* __restrict__ Xl,
    const __bf16* __restrict__ Wh, const __bf16* __restrict__ Wl,
    const float* __restrict__ bias, const int* __restrict__ par,
    const float* __restrict__ prev, int prevN,
    float* __restrict__ out, int Nl) {
  __shared__ __align__(16) char lds[2][BUF_BYTES];

  const int tid     = threadIdx.x;
  const int lane    = tid & 31;
  const int wave    = tid >> 5;
  const int rowBase = blockIdx.y * MT;
  const int colBase = blockIdx.x * NT;
  const int mOff    = (wave & 3) * 32;
  const int nOff    = (wave >> 2) * 32;

  // Stage one KT-slice (A hi/lo + B hi/lo) into an LDS buffer.
  auto stage = [&](char* buf, int kk) {
#pragma unroll
    for (int i = 0; i < CHUNKS_PER_THREAD; ++i) {
      int c = tid + i * 256;
      const __bf16* g;
      int loff;
      if (c < 1024) {  // A: 128 rows x 2 planes x 4 chunks
        int plane = c >> 9, r = (c >> 2) & 127, kc = c & 3;
        const __bf16* s = plane ? Xl : Xh;
        g    = s + (size_t)(rowBase + r) * 512 + kk + kc * 8;
        loff = plane * A_PLANE + r * A_ROW_STRIDE + kc * 16;
      } else {         // B: 64 cols x 2 planes x 4 chunks (K-major in ws)
        int c2 = c - 1024;
        int plane = c2 >> 8, col = (c2 >> 2) & 63, kc = c2 & 3;
        const __bf16* s = plane ? Wl : Wh;
        g    = s + (size_t)(colBase + col) * 512 + kk + kc * 8;
        loff = B_BASE + plane * B_PLANE + col * A_ROW_STRIDE + kc * 16;
      }
#if HAVE_ASYNC_LDS
      __builtin_amdgcn_global_load_async_to_lds_b128(
          (__attribute__((address_space(1))) v4i*)g,
          (__attribute__((address_space(3))) v4i*)(buf + loff),
          0, 0);
#else
      *(v4u*)(buf + loff) = *(const v4u*)g;
#endif
    }
  };

  v8f acc[2][2] = {};

  stage(lds[0], 0);
#pragma unroll 1
  for (int ks = 0; ks < NKSTEP; ++ks) {
    if (ks + 1 < NKSTEP) {
      stage(lds[(ks + 1) & 1], (ks + 1) * KT);
      wait_asynccnt_le<CHUNKS_PER_THREAD>();  // stage ks done (in-order)
    } else {
      wait_asynccnt_le<0>();
    }
    __syncthreads();

    const char* buf = lds[ks & 1];
    const int g = lane >> 4;

    Frag a[2][2];  // [mi][plane]
#pragma unroll
    for (int mi = 0; mi < 2; ++mi) {
      int r = mOff + mi * 16 + (lane & 15);
#pragma unroll
      for (int p = 0; p < 2; ++p) {
        const char* base = buf + p * A_PLANE + r * A_ROW_STRIDE;
        a[mi][p].q[0] = *(const v4u*)(base + g * 16);        // K g*8..g*8+7
        a[mi][p].q[1] = *(const v4u*)(base + 32 + g * 16);   // K 16+g*8..
      }
    }
    Frag b[2][2];  // [ni][plane]
#pragma unroll
    for (int ni = 0; ni < 2; ++ni) {
      int cc = nOff + ni * 16 + (lane & 15);
#pragma unroll
      for (int p = 0; p < 2; ++p) {
        const char* base = buf + B_BASE + p * B_PLANE + cc * A_ROW_STRIDE + g * 16;
        b[ni][p].q[0] = *(const v4u*)(base);                 // K g*8..g*8+7
        b[ni][p].q[1] = *(const v4u*)(base + 32);            // K 16+g*8..
      }
    }

#pragma unroll
    for (int mi = 0; mi < 2; ++mi)
#pragma unroll
      for (int ni = 0; ni < 2; ++ni) {
        acc[mi][ni] = __builtin_amdgcn_wmma_f32_16x16x32_bf16(
            false, a[mi][0].v, false, b[ni][0].v, (short)0, acc[mi][ni], false, false);
        acc[mi][ni] = __builtin_amdgcn_wmma_f32_16x16x32_bf16(
            false, a[mi][0].v, false, b[ni][1].v, (short)0, acc[mi][ni], false, false);
        acc[mi][ni] = __builtin_amdgcn_wmma_f32_16x16x32_bf16(
            false, a[mi][1].v, false, b[ni][0].v, (short)0, acc[mi][ni], false, false);
      }
    __syncthreads();
  }

  // Epilogue: bias + sigmoid + hierarchical gather-multiply + store.
#pragma unroll
  for (int ni = 0; ni < 2; ++ni) {
    int col = colBase + nOff + ni * 16 + (lane & 15);
    if (col < Nl) {
      float bb = bias[col];
      int pidx = 0;
      if (HASP) pidx = par[col];
#pragma unroll
      for (int mi = 0; mi < 2; ++mi) {
        int row0 = rowBase + mOff + mi * 16 + (lane >> 4) * 8;
#pragma unroll
        for (int r = 0; r < 8; ++r) {
          float logit = acc[mi][ni][r] + bb;
          float p = 1.0f / (1.0f + __expf(-logit));
          if (HASP) p *= prev[(size_t)(row0 + r) * prevN + pidx];
          out[(size_t)(row0 + r) * Nl + col] = p;
        }
      }
    }
  }
}

// ---------------------------------------------------------------------------
// Launch
// ---------------------------------------------------------------------------
extern "C" void kernel_launch(void* const* d_in, const int* in_sizes, int n_in,
                              void* d_out, int out_size, void* d_ws, size_t ws_size,
                              hipStream_t stream) {
  (void)in_sizes; (void)n_in; (void)out_size; (void)ws_size;

  const float* X = (const float*)d_in[0];
  const float* W[4]   = {(const float*)d_in[1], (const float*)d_in[3],
                         (const float*)d_in[5], (const float*)d_in[7]};
  const float* bv[4]  = {(const float*)d_in[2], (const float*)d_in[4],
                         (const float*)d_in[6], (const float*)d_in[8]};
  const int*   par[4] = {nullptr, (const int*)d_in[9],
                         (const int*)d_in[10], (const int*)d_in[11]};

  const int Nl[4]     = {20, 180, 1400, 18000};
  const int Npad[4]   = {64, 192, 1408, 18048};
  const int colOff[4] = {0, 64, 256, 1664};   // cumulative padded cols
  const size_t WCOLS  = 64 + 192 + 1408 + 18048;  // 19712

  char* ws = (char*)d_ws;
  const size_t XPB = (size_t)4096 * 512 * sizeof(__bf16);   // 4 MB / plane
  const size_t WPB = WCOLS * 512 * sizeof(__bf16);          // ~19.25 MB / plane

  __bf16* Xh    = (__bf16*)(ws);
  __bf16* Xl    = (__bf16*)(ws + XPB);
  __bf16* WhAll = (__bf16*)(ws + 2 * XPB);
  __bf16* WlAll = (__bf16*)(ws + 2 * XPB + WPB);
  float*  prob0 = (float*)(ws + 2 * XPB + 2 * WPB);
  float*  prob1 = prob0 + (size_t)4096 * 20;
  float*  prob2 = prob1 + (size_t)4096 * 180;

  // 1) split X to bf16 hi/lo
  {
    int n = 4096 * 512;
    hd_split_x<<<(n + 255) / 256, 256, 0, stream>>>(X, Xh, Xl, n);
  }
  // 2) split + transpose + pad each W
  for (int l = 0; l < 4; ++l) {
    dim3 grid(Npad[l] / 32, 512 / 32);
    dim3 blk(32, 8);
    hd_split_w<<<grid, blk, 0, stream>>>(W[l],
                                         WhAll + (size_t)colOff[l] * 512,
                                         WlAll + (size_t)colOff[l] * 512,
                                         Nl[l]);
  }
  // 3) fused level GEMMs
  float* outs[4]  = {prob0, prob1, prob2, (float*)d_out};
  const float* prevs[4] = {nullptr, prob0, prob1, prob2};
  const int prevN[4]    = {1, 20, 180, 1400};
  for (int l = 0; l < 4; ++l) {
    dim3 grid(Npad[l] / NT, 4096 / MT);
    const __bf16* wh = WhAll + (size_t)colOff[l] * 512;
    const __bf16* wl = WlAll + (size_t)colOff[l] * 512;
    if (l == 0) {
      hd_gemm<false><<<grid, 256, 0, stream>>>(Xh, Xl, wh, wl, bv[l], nullptr,
                                               nullptr, 1, outs[l], Nl[l]);
    } else {
      hd_gemm<true><<<grid, 256, 0, stream>>>(Xh, Xl, wh, wl, bv[l], par[l],
                                              prevs[l], prevN[l], outs[l], Nl[l]);
    }
  }
}
